// WMN_9844065042537
// MI455X (gfx1250) — matrix operand order repaired and checked by reading, not compiled, not run on hardware
//
#include <hip/hip_runtime.h>
#include <hip/hip_bf16.h>
#include <stdint.h>

typedef __attribute__((ext_vector_type(16))) _Float16 v16h;
typedef __attribute__((ext_vector_type(8)))  _Float16 v8h;
typedef __attribute__((ext_vector_type(8)))  float    v8f;

#define WMMA16(A,B,C) __builtin_amdgcn_wmma_f32_16x16x32_f16(false,(A),false,(B),(short)0,(C),false,false)

#define BB     64
#define MM     50
#define SS     20
#define SQL    20
#define EE     128
#define HH     256
#define G3     768
#define NVOC   50000
#define SEQEND 2

union FragU { v16h v; v8h h[2]; };

// A fragment (16x32 f16, MxK): lane L -> row m = L&15; elements 0..7 -> K = kb..kb+7,
// elements 8..15 -> K = 16+kb..16+kb+7, kb = 8*(L>>4). Source row-major [rows][ld].
__device__ __forceinline__ v16h frag_a(const _Float16* base, int ld, int row0, int k0) {
  int lane = threadIdx.x & 31;
  const _Float16* p = base + (size_t)(row0 + (lane & 15)) * ld + k0 + ((lane >> 4) << 3);
  FragU u;
  u.h[0] = *(const v8h*)(p);
  u.h[1] = *(const v8h*)(p + 16);
  return u.v;
}

// B fragment (32x16 f16, KxN) from W stored row-major [N][ld=K]: lane L -> col n = L&15;
// elements 0..7 -> K = kb..kb+7, elements 8..15 -> K = 16+kb..16+kb+7, kb = 8*(L>>4).
__device__ __forceinline__ v16h frag_b(const _Float16* base, int ld, int col0, int k0) {
  int lane = threadIdx.x & 31;
  const _Float16* p = base + (size_t)(col0 + (lane & 15)) * ld + k0 + ((lane >> 4) << 3);
  FragU u;
  u.h[0] = *(const v8h*)(p);
  u.h[1] = *(const v8h*)(p + 16);
  return u.v;
}

__device__ __forceinline__ float sigmoidf_(float x) { return 1.0f / (1.0f + __expf(-x)); }

// ---------------- small utility kernels ----------------

__global__ void k_f32_to_f16(const float* __restrict__ src, _Float16* __restrict__ dst, int n) {
  int i = blockIdx.x * 256 + threadIdx.x;
  if (i < n) dst[i] = (_Float16)src[i];
}

// Combined GRU weight: Wc[n][k], n in [0,768), k in [0,384): k<128 -> w_ih[n][k], else w_hh[n][k-128]
__global__ void k_build_wc(const float* __restrict__ w_ih, const float* __restrict__ w_hh,
                           _Float16* __restrict__ wc) {
  int i = blockIdx.x * 256 + threadIdx.x;
  if (i >= G3 * 384) return;
  int n = i / 384, k = i % 384;
  float v = (k < EE) ? w_ih[n * EE + k] : w_hh[n * HH + (k - EE)];
  wc[i] = (_Float16)v;
}

__global__ void k_eos(const int* __restrict__ tok, int nrows, int slen, int* __restrict__ idx) {
  int r = blockIdx.x * 256 + threadIdx.x;
  if (r >= nrows) return;
  int id = slen - 1;
  for (int s = 0; s < slen; ++s) {
    if (tok[r * slen + s] == SEQEND) { id = s; break; }
  }
  idx[r] = id;
}

// ---------------- persistent GRU kernel (one workgroup, WMMA per step) ----------------
// 512 threads = 16 waves. Wave w owns column-tile ct = w (16 h-cols) and all 4 row-tiles,
// so each loaded weight fragment feeds 4 WMMAs. 4 waves/SIMD leaves ~256 VGPRs/wave:
// 16 f32 accumulators (128 VGPRs) + 3 B-frags (24) + 1 A-frag (8) stay in registers.
// mode 0: story  (steps = B*S = 1280, rows = 50, hidden carried across b boundaries)
// mode 1: query  (steps = SQ  = 20,   rows = 64 = B)
__global__ __launch_bounds__(512)
void k_gru(const int* __restrict__ tokens, const _Float16* __restrict__ embed16,
           const _Float16* __restrict__ Wc,
           const float* __restrict__ b_ih, const float* __restrict__ b_hh,
           const int* __restrict__ outIdx, float* __restrict__ outBuf, int mode) {
  __shared__ _Float16 Xs[64 * EE];   // 16 KB  (rows >= `rows` zero-filled)
  __shared__ _Float16 Hs[64 * HH];   // 32 KB

  const int tid  = threadIdx.x;
  const int lane = tid & 31;
  const int ct   = tid >> 5;                 // 0..15: this wave's column tile
  const int c    = lane & 15;
  const int kb8  = (lane >> 4) << 3;
  const int steps = mode ? SQL : (BB * SS);
  const int rows  = mode ? 64 : MM;

  // gate biases for this wave's h-column
  const int ch = ct * 16 + c;
  const float bir  = b_ih[ch] + b_hh[ch];
  const float biz  = b_ih[HH + ch] + b_hh[HH + ch];
  const float bin_ = b_ih[2 * HH + ch];
  const float bhn_ = b_hh[2 * HH + ch];

  for (int i = tid; i < 64 * HH; i += 512) Hs[i] = (_Float16)0.0f;
  __syncthreads();

  int wofs = 0;   // always 0 at runtime; redefined each step by empty asm below so the
                  // compiler cannot hoist (and then spill) the t-invariant weight loads,
                  // while Wc itself stays traceable -> global_load path, not flat_load.

  for (int t = 0; t < steps; ++t) {
    const int bb = mode ? 0 : (t / SS);
    const int ss = mode ? t : (t % SS);

    // gather X_t = embed16[token] into LDS, 8 halfs (16B) per chunk
    for (int i = tid; i < 64 * 16; i += 512) {
      int m = i >> 4, c8 = (i & 15) << 3;
      v8h val = {};
      if (m < rows) {
        int tok = mode ? tokens[m * SQL + ss] : tokens[(bb * MM + m) * SS + ss];
        val = *(const v8h*)(embed16 + (size_t)tok * EE + c8);
      }
      *(v8h*)(Xs + m * EE + c8) = val;
    }
    __syncthreads();   // X ready; previous step's H writes visible

    asm volatile("" : "+s"(wofs));
    const _Float16* wcp = Wc + wofs;

    // gates = [X | H] @ [Wih | Whh]^T  (K = 384), tiles (rt=0..3, ct)
    v8f ar[4] = {}, az[4] = {}, ain[4] = {}, ahn[4] = {};
    #pragma unroll
    for (int kk = 0; kk < 12; ++kk) {
      const int k0 = kk * 32;
      const bool isx = (k0 < EE);
      v16h br = frag_b(wcp, 384, ct * 16, k0);
      v16h bz = frag_b(wcp, 384, HH + ct * 16, k0);
      v16h bn = frag_b(wcp, 384, 2 * HH + ct * 16, k0);
      #pragma unroll
      for (int rt = 0; rt < 4; ++rt) {
        v16h a = isx ? frag_a(Xs, EE, rt * 16, k0) : frag_a(Hs, HH, rt * 16, k0 - EE);
        ar[rt] = WMMA16(a, br, ar[rt]);
        az[rt] = WMMA16(a, bz, az[rt]);
        if (isx) ain[rt] = WMMA16(a, bn, ain[rt]);
        else     ahn[rt] = WMMA16(a, bn, ahn[rt]);
      }
    }
    __syncthreads();   // all reads of Hs complete before any h' write

    // gate nonlinearity + state update (each lane owns its (m, ch) cells)
    #pragma unroll
    for (int rt = 0; rt < 4; ++rt) {
      #pragma unroll
      for (int v = 0; v < 8; ++v) {
        int m = rt * 16 + v + kb8;
        float r  = sigmoidf_(ar[rt][v] + bir);
        float z  = sigmoidf_(az[rt][v] + biz);
        float nn = tanhf(ain[rt][v] + bin_ + r * (ahn[rt][v] + bhn_));
        float hp = (float)Hs[m * HH + ch];
        float h2 = (1.0f - z) * nn + z * hp;
        Hs[m * HH + ch] = (_Float16)h2;
        if (m < rows) {
          int orow = mode ? m : (bb * MM + m);
          if (outIdx[orow] == ss) outBuf[(size_t)orow * HH + ch] = h2;
        }
      }
    }
    // no barrier needed here: next gather only writes Xs; barrier after gather orders Hs
  }
}

// ---------------- generic WMMA GEMM: C[M,N] = act(A[M,K] @ B[N,K]^T + bias) ----------------
__global__ __launch_bounds__(256)
void k_gemm(const _Float16* __restrict__ A, const _Float16* __restrict__ B,
            const float* __restrict__ bias, float* __restrict__ Cf,
            _Float16* __restrict__ Ch, int M, int N, int K, int relu) {
  int gw = (blockIdx.x * 256 + threadIdx.x) >> 5;
  int lane = threadIdx.x & 31;
  int Nt = N >> 4;
  int tiles = (M >> 4) * Nt;
  if (gw >= tiles) return;            // wave-uniform: EXEC all-ones inside
  int mt = gw / Nt, nt = gw % Nt;
  v8f acc = {};
  for (int k0 = 0; k0 < K; k0 += 32) {
    v16h a = frag_a(A, K, mt << 4, k0);
    v16h b = frag_b(B, K, nt << 4, k0);
    acc = WMMA16(a, b, acc);
  }
  int n = (nt << 4) + (lane & 15);
  float bv = bias ? bias[n] : 0.0f;
  int mb = (mt << 4) + ((lane >> 4) << 3);
  #pragma unroll
  for (int v = 0; v < 8; ++v) {
    float x = acc[v] + bv;
    if (relu) x = fmaxf(x, 0.0f);
    size_t o = (size_t)(mb + v) * N + n;
    if (Cf) Cf[o] = x;
    if (Ch) Ch[o] = (_Float16)x;
  }
}

// ---------------- multi-hop attention (one block per batch element) ----------------
__global__ __launch_bounds__(256)
void k_hops(const float* __restrict__ mem_key, const float* __restrict__ mem,
            const float* __restrict__ rq,
            const float* __restrict__ ft_w1, const float* __restrict__ ft_b1,
            const float* __restrict__ ft_w2, const float* __restrict__ ft_b2,
            float* __restrict__ bufs) {
  const int b = blockIdx.x, tid = threadIdx.x;
  const int wave = tid >> 5, lane = tid & 31;
  __shared__ float ok[HH], at[64], buf[HH], tmp[HH];
  ok[tid] = rq[b * HH + tid];
  __syncthreads();
  for (int hop = 0; hop < 3; ++hop) {
    for (int m = wave; m < MM; m += 8) {
      float s = 0.0f;
      for (int j = lane; j < HH; j += 32) s += mem_key[((size_t)b * MM + m) * HH + j] * ok[j];
      for (int off = 16; off > 0; off >>= 1) s += __shfl_down(s, off, 32);
      if (lane == 0) at[m] = s * 0.0625f;   // 1/sqrt(256)
    }
    __syncthreads();
    if (tid == 0) {
      float mx = -1e30f;
      for (int m = 0; m < MM; ++m) mx = fmaxf(mx, at[m]);
      float sum = 0.0f;
      for (int m = 0; m < MM; ++m) { float e = __expf(at[m] - mx); at[m] = e; sum += e; }
      float inv = 1.0f / sum;
      for (int m = 0; m < MM; ++m) at[m] *= inv;
    }
    __syncthreads();
    float acc = 0.0f;
    for (int m = 0; m < MM; ++m) acc += at[m] * mem[((size_t)b * MM + m) * HH + tid];
    buf[tid] = acc;
    bufs[((size_t)b * 3 + hop) * HH + tid] = acc;
    __syncthreads();
    float t1 = ft_b1[tid];
    for (int k = 0; k < HH; ++k) t1 += buf[k] * ft_w1[tid * HH + k];
    tmp[tid] = fmaxf(t1, 0.0f);
    __syncthreads();
    float t2 = ft_b2[tid];
    for (int k = 0; k < HH; ++k) t2 += tmp[k] * ft_w2[tid * HH + k];
    __syncthreads();
    ok[tid] = t2;
    __syncthreads();
  }
}

// comb[b*9+p][0:256]=bufs[b][p%3]; [256:512]=bufs[b][p/3]; [512:768]=rq[b]  -> f16
__global__ void k_comb(const float* __restrict__ bufs, const float* __restrict__ rq,
                       _Float16* __restrict__ comb) {
  int i = blockIdx.x * 256 + threadIdx.x;
  if (i >= BB * 9 * G3) return;
  int row = i / G3, j = i % G3;
  int b = row / 9, p = row % 9;
  float v;
  if (j < HH)            v = bufs[((size_t)b * 3 + (p % 3)) * HH + j];
  else if (j < 2 * HH)   v = bufs[((size_t)b * 3 + (p / 3)) * HH + (j - HH)];
  else                   v = rq[(size_t)b * HH + (j - 2 * HH)];
  comb[i] = (_Float16)v;
}

// g16[b][j] = (f16) sum_{p<9} Y2[b*9+p][j]
__global__ void k_gsum(const float* __restrict__ Y2, _Float16* __restrict__ g16) {
  int i = blockIdx.x * 256 + threadIdx.x;
  if (i >= BB * G3) return;
  int b = i / G3, j = i % G3;
  float s = 0.0f;
  for (int p = 0; p < 9; ++p) s += Y2[((size_t)b * 9 + p) * G3 + j];
  g16[i] = (_Float16)s;
}

// ---------------- host side ----------------
extern "C" void kernel_launch(void* const* d_in, const int* in_sizes, int n_in,
                              void* d_out, int out_size, void* d_ws, size_t ws_size,
                              hipStream_t stream) {
  const int*   story = (const int*)d_in[0];
  const int*   query = (const int*)d_in[1];
  const float* embed = (const float*)d_in[2];
  const float* w_ih  = (const float*)d_in[3];
  const float* w_hh  = (const float*)d_in[4];
  const float* b_ih  = (const float*)d_in[5];
  const float* b_hh  = (const float*)d_in[6];
  const float* w_m   = (const float*)d_in[7];
  const float* ft_w1 = (const float*)d_in[8];
  const float* ft_b1 = (const float*)d_in[9];
  const float* ft_w2 = (const float*)d_in[10];
  const float* ft_b2 = (const float*)d_in[11];
  const float* g_w1  = (const float*)d_in[12];
  const float* g_b1  = (const float*)d_in[13];
  const float* g_w2  = (const float*)d_in[14];
  const float* g_b2  = (const float*)d_in[15];
  const float* f_w1  = (const float*)d_in[16];
  const float* f_b1  = (const float*)d_in[17];
  const float* f_w2  = (const float*)d_in[18];
  const float* f_b2  = (const float*)d_in[19];
  const float* v_w   = (const float*)d_in[20];
  float* out = (float*)d_out;

  size_t off = 0;
  auto alloc = [&](size_t bytes) -> void* {
    off = (off + 255) & ~(size_t)255;
    void* p = (char*)d_ws + off;
    off += bytes;
    return p;
  };
  _Float16* wc16   = (_Float16*)alloc((size_t)G3 * 384 * 2);
  _Float16* emb16  = (_Float16*)alloc((size_t)NVOC * EE * 2);
  _Float16* wm16   = (_Float16*)alloc((size_t)HH * HH * 2);
  _Float16* gw1_16 = (_Float16*)alloc((size_t)G3 * G3 * 2);
  _Float16* gw2_16 = (_Float16*)alloc((size_t)G3 * G3 * 2);
  _Float16* fw1_16 = (_Float16*)alloc((size_t)G3 * G3 * 2);
  _Float16* fw2_16 = (_Float16*)alloc((size_t)HH * G3 * 2);
  _Float16* vw16   = (_Float16*)alloc((size_t)NVOC * HH * 2);
  int*      eosIdx = (int*)alloc((size_t)BB * MM * 4);
  int*      qIdx   = (int*)alloc((size_t)BB * 4);
  float*    memf   = (float*)alloc((size_t)BB * MM * HH * 4);
  _Float16* mem16  = (_Float16*)alloc((size_t)BB * MM * HH * 2);
  float*    memkey = (float*)alloc((size_t)BB * MM * HH * 4);
  float*    rq     = (float*)alloc((size_t)BB * HH * 4);
  float*    bufs   = (float*)alloc((size_t)BB * 3 * HH * 4);
  _Float16* comb16 = (_Float16*)alloc((size_t)BB * 9 * G3 * 2);
  _Float16* Y1_16  = (_Float16*)alloc((size_t)BB * 9 * G3 * 2);
  float*    Y2_f32 = (float*)alloc((size_t)BB * 9 * G3 * 4);
  _Float16* g16    = (_Float16*)alloc((size_t)BB * G3 * 2);
  _Float16* R1_16  = (_Float16*)alloc((size_t)BB * G3 * 2);
  _Float16* rsn16  = (_Float16*)alloc((size_t)BB * HH * 2);

  auto cdiv = [](int a, int b) { return (a + b - 1) / b; };

  // weight / embedding conversions
  k_build_wc<<<cdiv(G3 * 384, 256), 256, 0, stream>>>(w_ih, w_hh, wc16);
  k_f32_to_f16<<<cdiv(NVOC * EE, 256), 256, 0, stream>>>(embed, emb16, NVOC * EE);
  k_f32_to_f16<<<cdiv(HH * HH, 256), 256, 0, stream>>>(w_m, wm16, HH * HH);
  k_f32_to_f16<<<cdiv(G3 * G3, 256), 256, 0, stream>>>(g_w1, gw1_16, G3 * G3);
  k_f32_to_f16<<<cdiv(G3 * G3, 256), 256, 0, stream>>>(g_w2, gw2_16, G3 * G3);
  k_f32_to_f16<<<cdiv(G3 * G3, 256), 256, 0, stream>>>(f_w1, fw1_16, G3 * G3);
  k_f32_to_f16<<<cdiv(HH * G3, 256), 256, 0, stream>>>(f_w2, fw2_16, HH * G3);
  k_f32_to_f16<<<cdiv(NVOC * HH, 256), 256, 0, stream>>>(v_w, vw16, NVOC * HH);

  // eos indices
  k_eos<<<cdiv(BB * MM, 256), 256, 0, stream>>>(story, BB * MM, SS, eosIdx);
  k_eos<<<1, 256, 0, stream>>>(query, BB, SQL, qIdx);

  // recurrent encoders (persistent single-WGP WMMA kernels)
  k_gru<<<1, 512, 0, stream>>>(story, emb16, wc16, b_ih, b_hh, eosIdx, memf, 0);
  k_gru<<<1, 512, 0, stream>>>(query, emb16, wc16, b_ih, b_hh, qIdx, rq, 1);

  // mem_key = memory_slots @ w_m^T
  k_f32_to_f16<<<cdiv(BB * MM * HH, 256), 256, 0, stream>>>(memf, mem16, BB * MM * HH);
  k_gemm<<<cdiv((BB * MM / 16) * (HH / 16), 8), 256, 0, stream>>>(
      mem16, wm16, nullptr, memkey, nullptr, BB * MM, HH, HH, 0);

  // multi-hop attention
  k_hops<<<BB, 256, 0, stream>>>(memkey, memf, rq, ft_w1, ft_b1, ft_w2, ft_b2, bufs);

  // relation module
  k_comb<<<cdiv(BB * 9 * G3, 256), 256, 0, stream>>>(bufs, rq, comb16);
  k_gemm<<<cdiv((BB * 9 / 16) * (G3 / 16), 8), 256, 0, stream>>>(
      comb16, gw1_16, g_b1, nullptr, Y1_16, BB * 9, G3, G3, 1);
  k_gemm<<<cdiv((BB * 9 / 16) * (G3 / 16), 8), 256, 0, stream>>>(
      Y1_16, gw2_16, g_b2, Y2_f32, nullptr, BB * 9, G3, G3, 1);
  k_gsum<<<cdiv(BB * G3, 256), 256, 0, stream>>>(Y2_f32, g16);
  k_gemm<<<cdiv((BB / 16) * (G3 / 16), 8), 256, 0, stream>>>(
      g16, fw1_16, f_b1, nullptr, R1_16, BB, G3, G3, 1);
  k_gemm<<<cdiv((BB / 16) * (HH / 16), 8), 256, 0, stream>>>(
      R1_16, fw2_16, f_b2, nullptr, rsn16, BB, HH, G3, 0);

  // final vocab projection (memory-bound: 25.6 MB f16 weights)
  k_gemm<<<cdiv((BB / 16) * (NVOC / 16), 8), 256, 0, stream>>>(
      rsn16, vw16, nullptr, out, nullptr, BB, NVOC, HH, 0);
}
